// VisionMamba_88725434401382
// MI455X (gfx1250) — compile-verified, matrix operand order checked by reading, not run on hardware
//
#include <hip/hip_runtime.h>
#include <hip/hip_bf16.h>
#include <math.h>
#include <stdint.h>

typedef __attribute__((ext_vector_type(16))) _Float16     v16h;
typedef __attribute__((ext_vector_type(8)))  float        v8f;
typedef __attribute__((ext_vector_type(4)))  unsigned int v4u;
typedef __attribute__((ext_vector_type(8)))  int          v8i;
typedef __attribute__((ext_vector_type(4)))  int          v4i;

#define CB    2
#define CT    16
#define CD    384
#define CDI   768
#define CDS   16
#define CDTR  24
#define CNH   8
#define CHD   48
#define CNC   1000
#define CHP   14
#define CNPF  196
#define CL    3137
#define CDEPTH 24
#define CROWS 6272     /* B*T*NPF */
#define CKP   768      /* CIN*P*P */
#define CQKV  1152
#define CXPN  56       /* DTR + 2*DS */

__device__ __forceinline__ float dev_silu(float x) { return x / (1.f + __expf(-x)); }

// ---------------------------------------------------------------------------
// Issue a TDM 2D tile load: 64 rows x 64 cols of f32 from row-major tensor
// (row stride = ldK elements) into LDS with 65-float row pitch (pad_enable,
// pad_interval=64 DW, pad_amount=1 DW). OOB rows/cols are zero-filled by TDM.
// ---------------------------------------------------------------------------
__device__ __forceinline__ void tdm_load_tile_f32(
    const float* gsrc, unsigned int lds_byte_addr, int kRem, int nRem, int ldK)
{
    unsigned long long ga = (unsigned long long)(size_t)gsrc;
    v4u g0;
    g0[0] = 1u;                                   // count=1 (valid user D#)
    g0[1] = lds_byte_addr;                        // lds_addr
    g0[2] = (unsigned int)(ga & 0xFFFFFFFFull);   // global_addr[31:0]
    g0[3] = (unsigned int)((ga >> 32) & 0x1FFFFFFull) | (2u << 30); // addr[56:32] | type=2

    v8i g1;
    g1[0] = (int)((2u << 16) | (1u << 20) | (5u << 22)); // data_size=4B, pad_en, interval=64DW (amount=1DW)
    g1[1] = (int)(((unsigned)kRem & 0xFFFFu) << 16);     // tensor_dim0[15:0]
    g1[2] = (int)((((unsigned)kRem >> 16) & 0xFFFFu) | (((unsigned)nRem & 0xFFFFu) << 16));
    g1[3] = (int)((((unsigned)nRem >> 16) & 0xFFFFu) | (64u << 16)); // tile_dim0=64
    g1[4] = 64;                                          // tile_dim1=64, tile_dim2=0
    g1[5] = ldK;                                         // tensor_dim0_stride[31:0]
    g1[6] = 0;
    g1[7] = 0;

    v4i gz = {0, 0, 0, 0};
#if __clang_major__ >= 23
    v8i gz8 = {0, 0, 0, 0, 0, 0, 0, 0};
    __builtin_amdgcn_tensor_load_to_lds(g0, g1, gz, gz, gz8, 0);
#else
    __builtin_amdgcn_tensor_load_to_lds(g0, g1, gz, gz, 0);
#endif
}

// ---------------------------------------------------------------------------
// Generic WMMA GEMM:  C[m,n] = epi( sum_k A[m,k] * W[n,k] + bias[n] )
// A via (aRS, aCS) strides; W row-major N x K (TDM-staged); C via (cRS, cCS).
// grid.z = batch. 64x64x64 tiles, 256 thr / 8 waves, 2 subtiles per wave.
// mode 0: +bias; mode 1: softplus(+bias); mode 2: resid + gamma*(x+bias)
// ---------------------------------------------------------------------------
__global__ __launch_bounds__(256) void k_gemm(
    const float* __restrict__ A, long long aBatch, long long aRS, long long aCS,
    const float* __restrict__ W,
    const float* __restrict__ bias,
    float* __restrict__ C, long long cBatch, long long cRS, long long cCS,
    const float* __restrict__ resid, long long rBatch, long long rRS,
    const float* __restrict__ gammaPtr,
    int M, int N, int K, int mode)
{
    __shared__ _Float16 As[64][72];   // padded f16 A tile
    __shared__ float    Wt[64][65];   // f32 W tile, TDM-written, pitch 65

    const int tid = threadIdx.x;
    const int m0  = blockIdx.y * 64;
    const int n0  = blockIdx.x * 64;
    const int bz  = blockIdx.z;

    A += (long long)bz * aBatch;
    C += (long long)bz * cBatch;
    if (resid) resid += (long long)bz * rBatch;

    const int lane = tid & 31;
    const int wave = tid >> 5;
    const int mi   = wave & 3;          // M subtile 0..3
    const int ni0  = (wave >> 2) * 2;   // N subtile pair base
    const int hlf  = lane >> 4;
    const int rlo  = lane & 15;

    const unsigned int wt_lds = (unsigned int)(size_t)&Wt[0][0];

    v8f acc0 = {0.f,0.f,0.f,0.f,0.f,0.f,0.f,0.f};
    v8f acc1 = acc0;

    for (int k0 = 0; k0 < K; k0 += 64) {
        // ---- async: wave 0 launches TDM for the weight tile ----
        if (tid < 32) {
            tdm_load_tile_f32(W + (long long)n0 * K + k0, wt_lds, K - k0, N - n0, K);
        }

        // ---- stage A tile as f16 (vectorized interior / branchless edge) ----
        const bool interior = (m0 + 64 <= M) && (k0 + 64 <= K);
        if (aCS == 1) {                       // A row-major: contiguous along k
            const int trow = tid >> 4;        // 0..15
            const int tk4  = (tid & 15) << 2; // 0..60
            if (interior) {
#pragma unroll
                for (int rr = 0; rr < 4; ++rr) {
                    int row = trow + rr * 16;
                    const float4 v = *reinterpret_cast<const float4*>(
                        A + (long long)(m0 + row) * aRS + (k0 + tk4));
                    As[row][tk4 + 0] = (_Float16)v.x;
                    As[row][tk4 + 1] = (_Float16)v.y;
                    As[row][tk4 + 2] = (_Float16)v.z;
                    As[row][tk4 + 3] = (_Float16)v.w;
                }
            } else {
#pragma unroll
                for (int rr = 0; rr < 4; ++rr) {
                    int row = trow + rr * 16;
                    int gm = m0 + row;
                    int gmc = gm < M ? gm : (M - 1);
#pragma unroll
                    for (int j = 0; j < 4; ++j) {
                        int gk = k0 + tk4 + j;
                        int gkc = gk < K ? gk : (K - 1);
                        float v = A[(long long)gmc * aRS + gkc];
                        As[row][tk4 + j] = (_Float16)((gm < M && gk < K) ? v : 0.f);
                    }
                }
            }
        } else {                              // A column-contiguous (aRS == 1)
            const int tm4 = (tid & 15) << 2;  // 4 consecutive m
            const int tkc = tid >> 4;
            if (interior) {
#pragma unroll
                for (int kr = 0; kr < 4; ++kr) {
                    int kk = tkc + kr * 16;
                    const float* p = A + (long long)(k0 + kk) * aCS + (m0 + tm4);
#pragma unroll
                    for (int j = 0; j < 4; ++j) As[tm4 + j][kk] = (_Float16)p[j];
                }
            } else {
#pragma unroll
                for (int kr = 0; kr < 4; ++kr) {
                    int kk = tkc + kr * 16;
                    int gk = k0 + kk;
                    int gkc = gk < K ? gk : (K - 1);
#pragma unroll
                    for (int j = 0; j < 4; ++j) {
                        int gm = m0 + tm4 + j;
                        int gmc = gm < M ? gm : (M - 1);
                        float v = A[(long long)gkc * aCS + gmc];
                        As[tm4 + j][kk] = (_Float16)((gm < M && gk < K) ? v : 0.f);
                    }
                }
            }
        }

        if (k0 + 64 < K) {  // gfx1250 global_prefetch_b8 on next A K-tile
            __builtin_prefetch(&A[(long long)(m0 + (tid & 63)) * aRS + (long long)(k0 + 64) * aCS], 0, 0);
        }

        if (tid < 32) __builtin_amdgcn_s_wait_tensorcnt(0);  // TDM tile landed
        __syncthreads();

        // ---- 2 k-substeps of WMMA per accumulator ----
#pragma unroll
        for (int ks = 0; ks < 64; ks += 32) {
            v16h fa, fb0, fb1;
#pragma unroll
            for (int e = 0; e < 16; ++e) {
                int kk = ks + e + 8 * ((e >= 8) + hlf);
                fa[e]  = As[mi * 16 + rlo][kk];
                fb0[e] = (_Float16)Wt[ni0 * 16 + rlo][kk];
                fb1[e] = (_Float16)Wt[(ni0 + 1) * 16 + rlo][kk];
            }
            acc0 = __builtin_amdgcn_wmma_f32_16x16x32_f16(false, fa, false, fb0, (short)0, acc0, false, false);
            acc1 = __builtin_amdgcn_wmma_f32_16x16x32_f16(false, fa, false, fb1, (short)0, acc1, false, false);
        }
        __syncthreads();
    }

    // ---- epilogue: C 16x16 f32 layout, VGPR r -> row r + 8*hlf, col rlo ----
    for (int t = 0; t < 2; ++t) {
        v8f acc = t ? acc1 : acc0;
        int gc = n0 + (ni0 + t) * 16 + rlo;
#pragma unroll
        for (int r = 0; r < 8; ++r) {
            int gr = m0 + mi * 16 + r + 8 * hlf;
            if (gr < M && gc < N) {
                float v = acc[r];
                if (bias) v += bias[gc];
                if (mode == 1)      v = (v > 20.f) ? v : log1pf(__expf(v));
                else if (mode == 2) v = resid[(long long)gr * rRS + gc] + gammaPtr[0] * v;
                C[(long long)gr * cRS + (long long)gc * cCS] = v;
            }
        }
    }
}

// ---------------------------------------------------------------------------
// im2col for patch embedding: Acol[row, k], row = bt*196 + patch, k = c*256+py*16+px
// ---------------------------------------------------------------------------
__global__ void k_im2col(const float* __restrict__ x, float* __restrict__ out, int total)
{
    int idx = blockIdx.x * 256 + threadIdx.x;
    if (idx >= total) return;
    int k   = idx % CKP;
    int row = idx / CKP;
    int bt = row / CNPF, n = row % CNPF;
    int b = bt / CT, t = bt % CT;
    int hp = n / CHP, wp = n % CHP;
    int c = k / 256, r = k % 256;
    int py = r / 16, px = r % 16;
    long long src = ((((long long)b * 3 + c) * CT + t) * 224 + (hp * 16 + py)) * 224 + (wp * 16 + px);
    out[idx] = x[src];
}

// ---------------------------------------------------------------------------
// Local grid attention: Manhattan-distance-1 neighbors only (<=4 per token)
// ---------------------------------------------------------------------------
__global__ void k_attn(const float* __restrict__ qkv, float* __restrict__ o)
{
    int idx = blockIdx.x * 256 + threadIdx.x;
    if (idx >= CROWS * CNH) return;
    int h   = idx & 7;
    int row = idx >> 3;             // bt*196 + n
    int n = row % CNPF;
    int rr = n / CHP, cc = n % CHP;
    int rowbase = row - n;

    const float* q = qkv + (long long)row * CQKV + h * CHD;

    int nb[4]; int cnt = 0;
    if (rr > 0)        nb[cnt++] = n - CHP;
    if (rr < CHP - 1)  nb[cnt++] = n + CHP;
    if (cc > 0)        nb[cnt++] = n - 1;
    if (cc < CHP - 1)  nb[cnt++] = n + 1;

    const float* vp[4];
    float s[4];
    float mx = -1e30f;
    for (int i = 0; i < cnt; ++i) {
        const float* kp = qkv + (long long)(rowbase + nb[i]) * CQKV + CD + h * CHD;
        vp[i] = qkv + (long long)(rowbase + nb[i]) * CQKV + 2 * CD + h * CHD;
        float d = 0.f;
        for (int j = 0; j < CHD; ++j) d += q[j] * kp[j];
        s[i] = d * 0.14433756729740643f;  // 1/sqrt(48)
        mx = fmaxf(mx, s[i]);
    }
    float sum = 0.f;
    for (int i = 0; i < cnt; ++i) { s[i] = __expf(s[i] - mx); sum += s[i]; }
    float inv = 1.f / sum;
    float* op = o + (long long)row * CD + h * CHD;
    for (int j = 0; j < CHD; ++j) {
        float a = 0.f;
        for (int i = 0; i < cnt; ++i) a += s[i] * vp[i][j];
        op[j] = a * inv;
    }
}

// ---------------------------------------------------------------------------
// Assemble hidden (B, L, D) with cls/pos/temporal embeds; zero residual
// ---------------------------------------------------------------------------
__global__ void k_assemble(const float* __restrict__ xr2, const float* __restrict__ cls,
                           const float* __restrict__ pos, const float* __restrict__ tpos,
                           float* __restrict__ H, float* __restrict__ RES, int total)
{
    int idx = blockIdx.x * 256 + threadIdx.x;
    if (idx >= total) return;
    int d = idx % CD;
    int rem = idx / CD;
    int p = rem % CL;
    int b = rem / CL;
    float v;
    if (p == 0) {
        v = cls[d] + pos[d];
    } else {
        int pp = p - 1;
        int t = pp / CNPF, n = pp % CNPF;
        v = xr2[(((long long)(b * CT + t) * CNPF) + n) * CD + d] + pos[(1 + n) * CD + d] + tpos[t * CD + d];
    }
    H[idx] = v;
    RES[idx] = 0.f;
}

// ---------------------------------------------------------------------------
// res += h; hn = rmsnorm(res) * w    (one 128-thread block per row of 384)
// ---------------------------------------------------------------------------
__global__ void k_rmsnorm(const float* __restrict__ H, float* __restrict__ RES,
                          const float* __restrict__ w, float* __restrict__ HN)
{
    __shared__ float red[128];
    int row = blockIdx.x;
    int tid = threadIdx.x;
    long long base = (long long)row * CD;
    float vals[3];
    float ss = 0.f;
#pragma unroll
    for (int i = 0; i < 3; ++i) {
        int d = tid + i * 128;
        float v = H[base + d] + RES[base + d];
        vals[i] = v;
        ss += v * v;
    }
    red[tid] = ss;
    __syncthreads();
    for (int s = 64; s > 0; s >>= 1) {
        if (tid < s) red[tid] += red[tid + s];
        __syncthreads();
    }
    float scale = rsqrtf(red[0] / (float)CD + 1e-5f);
#pragma unroll
    for (int i = 0; i < 3; ++i) {
        int d = tid + i * 128;
        RES[base + d] = vals[i];
        HN[base + d] = vals[i] * scale * w[d];
    }
}

// ---------------------------------------------------------------------------
// Depthwise causal conv (DC=4) + bias + silu, fwd and reversed branches
// ---------------------------------------------------------------------------
__global__ void k_conv(const float* __restrict__ XZ,
                       const float* __restrict__ cw, const float* __restrict__ cb,
                       const float* __restrict__ cwb, const float* __restrict__ cbb,
                       float* __restrict__ UCF, float* __restrict__ UCB, int total)
{
    int idx = blockIdx.x * 256 + threadIdx.x;
    if (idx >= total) return;
    int p = idx % CL;
    int rem = idx / CL;
    int d = rem % CDI;
    int b = rem / CDI;
    const float* xp = XZ + ((long long)b * 2 * CDI + d) * CL;
    float accf = cb[d];
#pragma unroll
    for (int j = 0; j < 4; ++j) {
        int q = p - 3 + j;
        if (q >= 0) accf += cw[d * 4 + j] * xp[q];
    }
    float accb = cbb[d];
#pragma unroll
    for (int j = 0; j < 4; ++j) {
        int q = p - 3 + j;
        if (q >= 0) accb += cwb[d * 4 + j] * xp[CL - 1 - q];
    }
    long long o = ((long long)b * CDI + d) * CL + p;
    UCF[o] = dev_silu(accf);
    UCB[o] = dev_silu(accb);
}

// ---------------------------------------------------------------------------
// Selective scan: 12 blocks (B x 2dirs x 3 chunks), one channel per lane,
// 16-entry state in registers. Gated with silu(z).
// ---------------------------------------------------------------------------
__global__ __launch_bounds__(256) void k_scan(
    const float* __restrict__ UCF, const float* __restrict__ UCB,
    const float* __restrict__ DTF, const float* __restrict__ DTB,
    const float* __restrict__ XDF, const float* __restrict__ XDB,
    const float* __restrict__ XZ,
    const float* __restrict__ alogf, const float* __restrict__ alogb,
    const float* __restrict__ dskf,  const float* __restrict__ dskb,
    float* __restrict__ YF, float* __restrict__ YB)
{
    int blk = blockIdx.x;
    int b = blk / 6;
    int rem = blk % 6;
    int dir = rem / 3;
    int chunk = rem % 3;
    int d = chunk * 256 + threadIdx.x;

    long long chbase = ((long long)b * CDI + d) * CL;
    const float* uc = (dir ? UCB : UCF) + chbase;
    const float* dt = (dir ? DTB : DTF) + chbase;
    const float* xd = (dir ? XDB : XDF) + (long long)b * CXPN * CL;
    const float* z  = XZ + ((long long)b * 2 * CDI + CDI + d) * CL;
    const float* al = (dir ? alogb : alogf) + d * CDS;
    float dsk = (dir ? dskb : dskf)[d];
    float* out = (dir ? YB : YF) + chbase;

    float Av[CDS], h[CDS];
#pragma unroll
    for (int s = 0; s < CDS; ++s) { Av[s] = -__expf(al[s]); h[s] = 0.f; }

    const float* Bp = xd + (long long)CDTR * CL;
    const float* Cp = xd + (long long)(CDTR + CDS) * CL;
    for (int p = 0; p < CL; ++p) {
        float dtv = dt[p];
        float uv  = uc[p];
        float du  = dtv * uv;
        float y = 0.f;
#pragma unroll
        for (int s = 0; s < CDS; ++s) {
            h[s] = __expf(dtv * Av[s]) * h[s] + du * Bp[(long long)s * CL + p];
            y += h[s] * Cp[(long long)s * CL + p];
        }
        int zp = dir ? (CL - 1 - p) : p;
        out[p] = (y + dsk * uv) * dev_silu(z[zp]);
    }
}

// ---------------------------------------------------------------------------
__global__ void k_combine(const float* __restrict__ YF, const float* __restrict__ YB,
                          float* __restrict__ YC, int total)
{
    int idx = blockIdx.x * 256 + threadIdx.x;
    if (idx >= total) return;
    int p = idx % CL;
    long long ch = idx / CL;
    YC[idx] = YF[idx] + YB[ch * CL + (CL - 1 - p)];
}

// ---------------------------------------------------------------------------
__global__ void k_finalnorm(const float* __restrict__ H, const float* __restrict__ RES,
                            const float* __restrict__ nfw, float* __restrict__ FV)
{
    __shared__ float red[128];
    int b = blockIdx.x;
    int tid = threadIdx.x;
    long long base = (long long)b * CL * CD;
    float vals[3];
    float ss = 0.f;
#pragma unroll
    for (int i = 0; i < 3; ++i) {
        int d = tid + i * 128;
        float v = H[base + d] + RES[base + d];
        vals[i] = v;
        ss += v * v;
    }
    red[tid] = ss;
    __syncthreads();
    for (int s = 64; s > 0; s >>= 1) {
        if (tid < s) red[tid] += red[tid + s];
        __syncthreads();
    }
    float scale = rsqrtf(red[0] / (float)CD + 1e-5f);
#pragma unroll
    for (int i = 0; i < 3; ++i) {
        int d = tid + i * 128;
        FV[b * CD + d] = vals[i] * scale * nfw[d];
    }
}

__global__ void k_head(const float* __restrict__ FV, const float* __restrict__ hw,
                       const float* __restrict__ hb, float* __restrict__ out)
{
    int idx = blockIdx.x * 256 + threadIdx.x;
    if (idx >= CB * CNC) return;
    int c = idx % CNC;
    int b = idx / CNC;
    float acc = hb[c];
    const float* f = FV + b * CD;
    const float* w = hw + (long long)c * CD;
    for (int d = 0; d < CD; ++d) acc += f[d] * w[d];
    out[idx] = acc;
}

// ---------------------------------------------------------------------------
extern "C" void kernel_launch(void* const* d_in, const int* in_sizes, int n_in,
                              void* d_out, int out_size, void* d_ws, size_t ws_size,
                              hipStream_t stream)
{
    (void)in_sizes; (void)n_in; (void)out_size; (void)ws_size;
    const float* x          = (const float*)d_in[0];
    const float* patch_w    = (const float*)d_in[1];
    const float* patch_b    = (const float*)d_in[2];
    const float* attn_in_w  = (const float*)d_in[3];
    const float* attn_in_b  = (const float*)d_in[4];
    const float* attn_out_w = (const float*)d_in[5];
    const float* attn_out_b = (const float*)d_in[6];
    const float* gamma      = (const float*)d_in[7];
    const float* cls_token  = (const float*)d_in[8];
    const float* pos_embed  = (const float*)d_in[9];
    const float* temporal   = (const float*)d_in[10];
    const float* norm_w     = (const float*)d_in[11];
    const float* in_proj_w  = (const float*)d_in[12];
    const float* conv_w     = (const float*)d_in[13];
    const float* conv_b     = (const float*)d_in[14];
    const float* xproj_w    = (const float*)d_in[15];
    const float* dtproj_w   = (const float*)d_in[16];
    const float* dtproj_b   = (const float*)d_in[17];
    const float* A_log      = (const float*)d_in[18];
    const float* D_skip     = (const float*)d_in[19];
    const float* conv_w_b   = (const float*)d_in[20];
    const float* conv_b_b   = (const float*)d_in[21];
    const float* xproj_w_b  = (const float*)d_in[22];
    const float* dtproj_w_b = (const float*)d_in[23];
    const float* dtproj_b_b = (const float*)d_in[24];
    const float* A_log_b    = (const float*)d_in[25];
    const float* D_skip_b   = (const float*)d_in[26];
    const float* out_proj_w = (const float*)d_in[27];
    const float* norm_f_w   = (const float*)d_in[28];
    const float* head_w     = (const float*)d_in[29];
    const float* head_b     = (const float*)d_in[30];
    float* out = (float*)d_out;
    float* ws  = (float*)d_ws;

    const long long HL = (long long)CB * CL * CD;
    const long long OFF_H   = 0;
    const long long OFF_RES = OFF_H + HL;
    const long long OFF_HN  = OFF_RES + HL;
    const long long PH      = OFF_HN + HL;

    // phase-1 pool
    const long long OFF_XR  = PH;
    const long long OFF_XR2 = OFF_XR  + (long long)CROWS * CD;
    const long long OFF_O   = OFF_XR2 + (long long)CROWS * CD;
    const long long OFF_QKV = OFF_O   + (long long)CROWS * CD;
    const long long OFF_COL = OFF_QKV + (long long)CROWS * CQKV;

    // phase-2 pool (aliases phase-1, used after k_assemble)
    const long long SZ_XZ = (long long)CB * 2 * CDI * CL;
    const long long SZ_UC = (long long)CB * CDI * CL;
    const long long SZ_XD = (long long)CB * CXPN * CL;
    const long long OFF_XZ  = PH;
    const long long OFF_UCF = OFF_XZ  + SZ_XZ;
    const long long OFF_UCB = OFF_UCF + SZ_UC;
    const long long OFF_XDF = OFF_UCB + SZ_UC;
    const long long OFF_XDB = OFF_XDF + SZ_XD;
    const long long OFF_DTF = OFF_XDB + SZ_XD;
    const long long OFF_DTB = OFF_DTF + SZ_UC;
    const long long OFF_YF  = OFF_DTB + SZ_UC;
    const long long OFF_YB  = OFF_YF  + SZ_UC;
    const long long OFF_YC  = OFF_YB  + SZ_UC;
    const long long OFF_FV  = OFF_YC  + SZ_UC;

    dim3 blk256(256);

    // ---- phase 1: patch embed + local attention ----
    {
        int tot = CROWS * CKP;
        k_im2col<<<dim3((tot + 255) / 256), blk256, 0, stream>>>(x, ws + OFF_COL, tot);
    }
    k_gemm<<<dim3((CD + 63) / 64, (CROWS + 63) / 64, 1), blk256, 0, stream>>>(
        ws + OFF_COL, 0, CKP, 1, patch_w, patch_b,
        ws + OFF_XR, 0, CD, 1, nullptr, 0, 0, nullptr, CROWS, CD, CKP, 0);
    k_gemm<<<dim3((CQKV + 63) / 64, (CROWS + 63) / 64, 1), blk256, 0, stream>>>(
        ws + OFF_XR, 0, CD, 1, attn_in_w, attn_in_b,
        ws + OFF_QKV, 0, CQKV, 1, nullptr, 0, 0, nullptr, CROWS, CQKV, CD, 0);
    k_attn<<<dim3((CROWS * CNH + 255) / 256), blk256, 0, stream>>>(ws + OFF_QKV, ws + OFF_O);
    k_gemm<<<dim3((CD + 63) / 64, (CROWS + 63) / 64, 1), blk256, 0, stream>>>(
        ws + OFF_O, 0, CD, 1, attn_out_w, attn_out_b,
        ws + OFF_XR2, 0, CD, 1, ws + OFF_XR, 0, CD, gamma, CROWS, CD, CD, 2);
    {
        int tot = (int)HL;
        k_assemble<<<dim3((tot + 255) / 256), blk256, 0, stream>>>(
            ws + OFF_XR2, cls_token, pos_embed, temporal, ws + OFF_H, ws + OFF_RES, tot);
    }

    // ---- phase 2: 24 Mamba layers ----
    for (int l = 0; l < CDEPTH; ++l) {
        const float* nw   = norm_w     + (long long)l * CD;
        const float* ipw  = in_proj_w  + (long long)l * 2 * CDI * CD;
        const float* cw   = conv_w     + (long long)l * CDI * 4;
        const float* cb   = conv_b     + (long long)l * CDI;
        const float* xw   = xproj_w    + (long long)l * CXPN * CDI;
        const float* dtw  = dtproj_w   + (long long)l * CDI * CDTR;
        const float* dtb  = dtproj_b   + (long long)l * CDI;
        const float* al   = A_log      + (long long)l * CDI * CDS;
        const float* dsk  = D_skip     + (long long)l * CDI;
        const float* cwb  = conv_w_b   + (long long)l * CDI * 4;
        const float* cbb  = conv_b_b   + (long long)l * CDI;
        const float* xwb  = xproj_w_b  + (long long)l * CXPN * CDI;
        const float* dtwb = dtproj_w_b + (long long)l * CDI * CDTR;
        const float* dtbb = dtproj_b_b + (long long)l * CDI;
        const float* alb  = A_log_b    + (long long)l * CDI * CDS;
        const float* dskb = D_skip_b   + (long long)l * CDI;
        const float* opw  = out_proj_w + (long long)l * CD * CDI;

        k_rmsnorm<<<dim3(CB * CL), dim3(128), 0, stream>>>(ws + OFF_H, ws + OFF_RES, nw, ws + OFF_HN);

        k_gemm<<<dim3((2 * CDI + 63) / 64, (CL + 63) / 64, CB), blk256, 0, stream>>>(
            ws + OFF_HN, (long long)CL * CD, CD, 1, ipw, nullptr,
            ws + OFF_XZ, (long long)2 * CDI * CL, 1, CL, nullptr, 0, 0, nullptr, CL, 2 * CDI, CD, 0);

        {
            int tot = CB * CDI * CL;
            k_conv<<<dim3((tot + 255) / 256), blk256, 0, stream>>>(
                ws + OFF_XZ, cw, cb, cwb, cbb, ws + OFF_UCF, ws + OFF_UCB, tot);
        }

        k_gemm<<<dim3(1, (CL + 63) / 64, CB), blk256, 0, stream>>>(
            ws + OFF_UCF, SZ_UC, 1, CL, xw, nullptr,
            ws + OFF_XDF, SZ_XD, 1, CL, nullptr, 0, 0, nullptr, CL, CXPN, CDI, 0);
        k_gemm<<<dim3(1, (CL + 63) / 64, CB), blk256, 0, stream>>>(
            ws + OFF_UCB, SZ_UC, 1, CL, xwb, nullptr,
            ws + OFF_XDB, SZ_XD, 1, CL, nullptr, 0, 0, nullptr, CL, CXPN, CDI, 0);

        k_gemm<<<dim3((CDI + 63) / 64, (CL + 63) / 64, CB), blk256, 0, stream>>>(
            ws + OFF_XDF, SZ_XD, 1, CL, dtw, dtb,
            ws + OFF_DTF, SZ_UC, 1, CL, nullptr, 0, 0, nullptr, CL, CDI, CDTR, 1);
        k_gemm<<<dim3((CDI + 63) / 64, (CL + 63) / 64, CB), blk256, 0, stream>>>(
            ws + OFF_XDB, SZ_XD, 1, CL, dtwb, dtbb,
            ws + OFF_DTB, SZ_UC, 1, CL, nullptr, 0, 0, nullptr, CL, CDI, CDTR, 1);

        k_scan<<<dim3(CB * 2 * (CDI / 256)), blk256, 0, stream>>>(
            ws + OFF_UCF, ws + OFF_UCB, ws + OFF_DTF, ws + OFF_DTB,
            ws + OFF_XDF, ws + OFF_XDB, ws + OFF_XZ,
            al, alb, dsk, dskb, ws + OFF_YF, ws + OFF_YB);

        {
            int tot = CB * CDI * CL;
            k_combine<<<dim3((tot + 255) / 256), blk256, 0, stream>>>(
                ws + OFF_YF, ws + OFF_YB, ws + OFF_YC, tot);
        }

        k_gemm<<<dim3((CD + 63) / 64, (CL + 63) / 64, CB), blk256, 0, stream>>>(
            ws + OFF_YC, SZ_UC, 1, CL, opw, nullptr,
            ws + OFF_H, (long long)CL * CD, CD, 1, nullptr, 0, 0, nullptr, CL, CD, CDI, 0);
    }

    // ---- head (only position 0 is used) ----
    k_finalnorm<<<dim3(CB), dim3(128), 0, stream>>>(ws + OFF_H, ws + OFF_RES, norm_f_w, ws + OFF_FV);
    k_head<<<dim3((CB * CNC + 255) / 256), blk256, 0, stream>>>(ws + OFF_FV, head_w, head_b, out);
}
